// EnhancedMemoryAttentionLayer_64020782514284
// MI455X (gfx1250) — compile-verified, matrix operand order checked by reading, not run on hardware
//
#include <hip/hip_runtime.h>

// ---------------------------------------------------------------------------
// CDNA5 (gfx1250) bf16-WMMA implementation of a DETR decoder layer.
// L=M=4096, B=1, D=256, H=8, DH=32, F=2048.  wave32; V_WMMA_F32_16X16X32_BF16.
// ---------------------------------------------------------------------------

typedef __bf16 bf16_t;
typedef __attribute__((ext_vector_type(8)))  __bf16 v8bf;
typedef __attribute__((ext_vector_type(16))) __bf16 v16bf;
typedef __attribute__((ext_vector_type(8)))  float  v8f;
typedef __attribute__((ext_vector_type(4)))  int    v4i;

#define LL 4096
#define DD 256
#define HH 8
#define MM 4096
#define FF 2048
#define DHH 32

#if __has_builtin(__builtin_amdgcn_global_load_async_to_lds_b128)
#define HAVE_ASYNC_LDS 1
typedef __attribute__((address_space(1))) v4i* g4p;   // global int4*
typedef __attribute__((address_space(3))) v4i* l4p;   // LDS int4*
#else
#define HAVE_ASYNC_LDS 0
#endif

__device__ __forceinline__ void wait_async0() {
#if __has_builtin(__builtin_amdgcn_s_wait_asynccnt)
    __builtin_amdgcn_s_wait_asynccnt(0);
#else
    asm volatile("s_wait_asynccnt 0x0" ::: "memory");
#endif
}

__device__ __forceinline__ bf16_t f2bf(float f) {
    unsigned int u = __float_as_uint(f);
    unsigned int r = u + 0x7fffu + ((u >> 16) & 1u);   // round-to-nearest-even
    unsigned short s = (unsigned short)(r >> 16);
    union { unsigned short u16; bf16_t b; } cv; cv.u16 = s; return cv.b;
}

__device__ __forceinline__ v8f wmma_bf16(v16bf a, v16bf b, v8f c) {
    return __builtin_amdgcn_wmma_f32_16x16x32_bf16(false, a, false, b,
                                                   (short)0, c, false, false);
}

// Load a 16-bit A/B fragment (16x32 / 32x16) for one wave32 lane from a
// row-major bf16 row pointer: two contiguous 8-element (16B) runs.
__device__ __forceinline__ v16bf load_frag_row(const bf16_t* rowp, int half) {
    v16bf v;
    const v8bf lo = *(const v8bf*)(rowp + half * 8);
    const v8bf hi = *(const v8bf*)(rowp + 16 + half * 8);
#pragma unroll
    for (int i = 0; i < 8; ++i) { v[i] = lo[i]; v[i + 8] = hi[i]; }
    return v;
}

// ------------------------------ prep kernels -------------------------------

__global__ void k_prep(const float* __restrict__ tgt, const float* __restrict__ qpos,
                       const float* __restrict__ mem, const float* __restrict__ pos,
                       bf16_t* __restrict__ qself, bf16_t* __restrict__ tgtb,
                       bf16_t* __restrict__ memk,  bf16_t* __restrict__ memb, int n) {
    int i = blockIdx.x * blockDim.x + threadIdx.x;
    if (i < n) {
        float t = tgt[i];
        qself[i] = f2bf(t + qpos[i]);
        tgtb[i]  = f2bf(t);
        float m = mem[i];
        memk[i]  = f2bf(m + pos[i]);
        memb[i]  = f2bf(m);
    }
}

__global__ void k_cvt(const float* __restrict__ src, bf16_t* __restrict__ dst, int n) {
    int i = blockIdx.x * blockDim.x + threadIdx.x;
    if (i < n) dst[i] = f2bf(src[i]);
}

// ------------------------------ WMMA GEMM ----------------------------------
// C[Md,Nd] = A[Md,Kd](bf16) @ W[Nd,Kd]^T(bf16) + bias[Nd]; optional ReLU.
// One wave (32 threads) per 32x64 output tile: 2 A-frags x 4 B-frags =
// 8 WMMA per K-step (K-step = 32), so each B fragment is reused twice and the
// scheduler has 8 independent WMMAs to overlap with the global loads.

__global__ __launch_bounds__(32)
void k_gemm(const bf16_t* __restrict__ A, const bf16_t* __restrict__ W,
            const float* __restrict__ bias, float* __restrict__ Cf,
            bf16_t* __restrict__ Cb, int Md, int Nd, int Kd, int relu) {
    const int lane = threadIdx.x;
    const int half = lane >> 4;
    const int ln   = lane & 15;
    const int tm = blockIdx.x, tn = blockIdx.y;

    const v8f z = {0.f, 0.f, 0.f, 0.f, 0.f, 0.f, 0.f, 0.f};
    v8f acc[2][4] = {{z, z, z, z}, {z, z, z, z}};

    const bf16_t* arow0 = A + (size_t)(tm * 32 + ln)      * Kd;
    const bf16_t* arow1 = A + (size_t)(tm * 32 + 16 + ln) * Kd;
    for (int k0 = 0; k0 < Kd; k0 += 32) {
        const v16bf a0 = load_frag_row(arow0 + k0, half);
        const v16bf a1 = load_frag_row(arow1 + k0, half);
#pragma unroll
        for (int f = 0; f < 4; ++f) {
            const int col = tn * 64 + f * 16 + ln;
            const v16bf b = load_frag_row(W + (size_t)col * Kd + k0, half);
            acc[0][f] = wmma_bf16(a0, b, acc[0][f]);
            acc[1][f] = wmma_bf16(a1, b, acc[1][f]);
        }
    }
#pragma unroll
    for (int g = 0; g < 2; ++g) {
#pragma unroll
        for (int f = 0; f < 4; ++f) {
            const int col = tn * 64 + f * 16 + ln;
            const float bs = bias ? bias[col] : 0.f;
#pragma unroll
            for (int r = 0; r < 8; ++r) {
                const int m = tm * 32 + g * 16 + r + half * 8;
                float v = acc[g][f][r] + bs;
                if (relu) v = fmaxf(v, 0.f);
                const size_t idx = (size_t)m * Nd + col;
                if (Cf) Cf[idx] = v;
                if (Cb) Cb[idx] = f2bf(v);
            }
        }
    }
}

// --------------------------- flash attention -------------------------------
// One wave per (16-query tile, head). Streams 32-key chunks:
//   V chunk staged memory->LDS via GLOBAL_LOAD_ASYNC_TO_LDS_B128 (ASYNCcnt),
//   S = Q Kt (2 WMMA), online softmax (shfl_xor over 16-lane half),
//   P transposed via LDS, O += P V (2 WMMA).

__global__ __launch_bounds__(32)
void k_attn(const bf16_t* __restrict__ Q, const bf16_t* __restrict__ K,
            const bf16_t* __restrict__ V, bf16_t* __restrict__ O,
            int Lk, float scale) {
    __shared__ bf16_t Pt[16 * 32];   // P tile, [query][key-in-chunk]
    __shared__ bf16_t Vt[32 * 32];   // V chunk, [key-in-chunk][dh]

    const int lane = threadIdx.x;
    const int half = lane >> 4;
    const int ln   = lane & 15;
    const int head = blockIdx.y;
    const int tq   = blockIdx.x;
    const int qrow = tq * 16 + ln;

    const v16bf aq = load_frag_row(Q + (size_t)qrow * DD + head * DHH, half);

    const v8f z = {0.f, 0.f, 0.f, 0.f, 0.f, 0.f, 0.f, 0.f};
    v8f acc0 = z, acc1 = z;
    float rmax[8], rsum[8];
#pragma unroll
    for (int r = 0; r < 8; ++r) { rmax[r] = -3.0e38f; rsum[r] = 0.f; }

    for (int jc = 0; jc < Lk; jc += 32) {
        // stage V chunk: lane -> key jc+lane, 32 contiguous bf16 (64B)
        {
            const bf16_t* vsrc = V + (size_t)(jc + lane) * DD + head * DHH;
            bf16_t* vdst = Vt + lane * 32;
#if HAVE_ASYNC_LDS
            g4p g = (g4p)(unsigned long long)vsrc;
            l4p l = (l4p)(unsigned)(unsigned long long)vdst;
            __builtin_amdgcn_global_load_async_to_lds_b128(g, l, 0, 0);
            __builtin_amdgcn_global_load_async_to_lds_b128(g, l, 16, 0);
            __builtin_amdgcn_global_load_async_to_lds_b128(g, l, 32, 0);
            __builtin_amdgcn_global_load_async_to_lds_b128(g, l, 48, 0);
#else
            const v8bf* src = (const v8bf*)vsrc;
            v8bf* dst = (v8bf*)vdst;
            dst[0] = src[0]; dst[1] = src[1]; dst[2] = src[2]; dst[3] = src[3];
#endif
        }
        // prefetch next chunk's K rows while we compute this one
        if (jc + 32 < Lk)
            __builtin_prefetch(K + (size_t)(jc + 32 + lane) * DD + head * DHH, 0, 1);

        // K fragments straight from global (rows are contiguous per token)
        const v16bf bk0 = load_frag_row(K + (size_t)(jc + ln)      * DD + head * DHH, half);
        const v16bf bk1 = load_frag_row(K + (size_t)(jc + 16 + ln) * DD + head * DHH, half);

        v8f s0 = wmma_bf16(aq, bk0, z);
        v8f s1 = wmma_bf16(aq, bk1, z);
        __syncthreads();

        // online softmax; query of (r,half) identical in S and O fragments
#pragma unroll
        for (int r = 0; r < 8; ++r) {
            float x0 = s0[r] * scale, x1 = s1[r] * scale;
            float mx = fmaxf(x0, x1);
#pragma unroll
            for (int o = 1; o < 16; o <<= 1) mx = fmaxf(mx, __shfl_xor(mx, o, 32));
            const float nm   = fmaxf(rmax[r], mx);
            const float corr = __expf(rmax[r] - nm);
            rmax[r] = nm;
            const float p0 = __expf(x0 - nm), p1 = __expf(x1 - nm);
            float cs = p0 + p1;
#pragma unroll
            for (int o = 1; o < 16; o <<= 1) cs += __shfl_xor(cs, o, 32);
            rsum[r] = rsum[r] * corr + cs;
            acc0[r] *= corr; acc1[r] *= corr;
            const int q = r + half * 8;
            Pt[q * 32 + ln]      = f2bf(p0);
            Pt[q * 32 + 16 + ln] = f2bf(p1);
        }
        __syncthreads();

        // P as A-fragment (rows contiguous in LDS)
        const v16bf ap = load_frag_row(Pt + ln * 32, half);

#if HAVE_ASYNC_LDS
        wait_async0();               // V chunk resident in LDS before gather
#endif
        // V as two B-fragments (gather, stride 32 in LDS)
        v16bf bv0, bv1;
#pragma unroll
        for (int i = 0; i < 8; ++i) {
            const int k1 = half * 8 + i;
            const int k2 = 16 + half * 8 + i;
            bv0[i]     = Vt[k1 * 32 + ln];
            bv1[i]     = Vt[k1 * 32 + 16 + ln];
            bv0[i + 8] = Vt[k2 * 32 + ln];
            bv1[i + 8] = Vt[k2 * 32 + 16 + ln];
        }
        acc0 = wmma_bf16(ap, bv0, acc0);
        acc1 = wmma_bf16(ap, bv1, acc1);
        __syncthreads();
    }

#pragma unroll
    for (int r = 0; r < 8; ++r) {
        const int q = tq * 16 + r + half * 8;
        const float inv = 1.0f / rsum[r];
        bf16_t* op = O + (size_t)q * DD + head * DHH;
        op[ln]      = f2bf(acc0[r] * inv);
        op[16 + ln] = f2bf(acc1[r] * inv);
    }
}

// ------------------------- fused residual + LayerNorm ----------------------
// One 256-thread block per row (D=256). Optional bf16 copy and optional
// bf16(y + extra) second output (cross-attention query input).

__global__ __launch_bounds__(256)
void k_add_ln(const float* __restrict__ res, const float* __restrict__ x,
              const float* __restrict__ g, const float* __restrict__ b,
              const float* __restrict__ extra, float* __restrict__ outf,
              bf16_t* __restrict__ outb, bf16_t* __restrict__ out2b) {
    const int row = blockIdx.x;
    const int t = threadIdx.x;
    const size_t idx = (size_t)row * DD + t;
    const float v = res[idx] + x[idx];

    __shared__ float red[8];
    __shared__ float mu_s, ri_s;

    float s = v;
#pragma unroll
    for (int o = 16; o > 0; o >>= 1) s += __shfl_xor(s, o, 32);
    if ((t & 31) == 0) red[t >> 5] = s;
    __syncthreads();
    if (t == 0) {
        float a = 0.f;
        for (int i = 0; i < 8; ++i) a += red[i];
        mu_s = a * (1.0f / 256.0f);
    }
    __syncthreads();
    const float mu = mu_s;
    const float d = v - mu;
    float s2 = d * d;
#pragma unroll
    for (int o = 16; o > 0; o >>= 1) s2 += __shfl_xor(s2, o, 32);
    if ((t & 31) == 0) red[t >> 5] = s2;
    __syncthreads();
    if (t == 0) {
        float a = 0.f;
        for (int i = 0; i < 8; ++i) a += red[i];
        ri_s = rsqrtf(a * (1.0f / 256.0f) + 1e-5f);
    }
    __syncthreads();
    const float y = d * ri_s * g[t] + b[t];
    if (outf)  outf[idx]  = y;
    if (outb)  outb[idx]  = f2bf(y);
    if (out2b) out2b[idx] = f2bf(y + extra[idx]);
}

// ------------------------------- launcher ----------------------------------

extern "C" void kernel_launch(void* const* d_in, const int* in_sizes, int n_in,
                              void* d_out, int out_size, void* d_ws, size_t ws_size,
                              hipStream_t stream) {
    (void)in_sizes; (void)n_in; (void)out_size; (void)ws_size;

    const float* tgt    = (const float*)d_in[0];
    const float* memory = (const float*)d_in[1];
    const float* pos    = (const float*)d_in[2];
    const float* qpos   = (const float*)d_in[3];
    const float* w_in_s = (const float*)d_in[4];
    const float* b_in_s = (const float*)d_in[5];
    const float* w_out_s= (const float*)d_in[6];
    const float* b_out_s= (const float*)d_in[7];
    const float* w_in_c = (const float*)d_in[8];
    const float* b_in_c = (const float*)d_in[9];
    const float* w_out_c= (const float*)d_in[10];
    const float* b_out_c= (const float*)d_in[11];
    const float* w1     = (const float*)d_in[12];
    const float* b1     = (const float*)d_in[13];
    const float* w2     = (const float*)d_in[14];
    const float* b2     = (const float*)d_in[15];
    const float* g1     = (const float*)d_in[16];
    const float* be1    = (const float*)d_in[17];
    const float* g2     = (const float*)d_in[18];
    const float* be2    = (const float*)d_in[19];
    const float* g3     = (const float*)d_in[20];
    const float* be3    = (const float*)d_in[21];

    const size_t LD = (size_t)LL * DD;
    const size_t MD = (size_t)MM * DD;
    const size_t LF = (size_t)LL * FF;

    char* wp = (char*)d_ws;
    auto alloc = [&](size_t bytes) -> void* {
        void* p = (void*)wp;
        wp += (bytes + 255) & ~(size_t)255;
        return p;
    };

    // bf16 buffers (several reused across sequential stages)
    bf16_t* qx  = (bf16_t*)alloc(LD * 2);          // tgt+qpos, later ln1+qpos
    bf16_t* tg  = (bf16_t*)alloc(LD * 2);          // bf16(tgt), later bf16(ln2)
    bf16_t* mk  = (bf16_t*)alloc(MD * 2);          // memory+pos
    bf16_t* mm  = (bf16_t*)alloc(MD * 2);          // memory
    bf16_t* Qb  = (bf16_t*)alloc(LD * 2);
    bf16_t* Kb  = (bf16_t*)alloc(MD * 2);
    bf16_t* Vb  = (bf16_t*)alloc(MD * 2);
    bf16_t* at  = (bf16_t*)alloc(LD * 2);          // attention output (both attns)
    bf16_t* ffh = (bf16_t*)alloc(LF * 2);          // FFN hidden
    bf16_t* wisb= (bf16_t*)alloc((size_t)3*DD*DD * 2);
    bf16_t* wosb= (bf16_t*)alloc((size_t)DD*DD * 2);
    bf16_t* wicb= (bf16_t*)alloc((size_t)3*DD*DD * 2);
    bf16_t* wocb= (bf16_t*)alloc((size_t)DD*DD * 2);
    bf16_t* w1b = (bf16_t*)alloc((size_t)FF*DD * 2);
    bf16_t* w2b = (bf16_t*)alloc((size_t)DD*FF * 2);
    // fp32 buffers
    float* t2   = (float*)alloc(LD * 4);           // GEMM fp32 out (reused 3x)
    float* ln1f = (float*)alloc(LD * 4);
    float* ln2f = (float*)alloc(LD * 4);

    const dim3 blk256(256), blk32(32);

    // --- prep: bf16 activations and weights ---
    k_prep<<<dim3((unsigned)(LD / 256)), blk256, 0, stream>>>(
        tgt, qpos, memory, pos, qx, tg, mk, mm, (int)LD);
    k_cvt<<<dim3((3*DD*DD + 255) / 256), blk256, 0, stream>>>(w_in_s, wisb, 3*DD*DD);
    k_cvt<<<dim3((DD*DD   + 255) / 256), blk256, 0, stream>>>(w_out_s, wosb, DD*DD);
    k_cvt<<<dim3((3*DD*DD + 255) / 256), blk256, 0, stream>>>(w_in_c, wicb, 3*DD*DD);
    k_cvt<<<dim3((DD*DD   + 255) / 256), blk256, 0, stream>>>(w_out_c, wocb, DD*DD);
    k_cvt<<<dim3((FF*DD   + 255) / 256), blk256, 0, stream>>>(w1, w1b, FF*DD);
    k_cvt<<<dim3((DD*FF   + 255) / 256), blk256, 0, stream>>>(w2, w2b, DD*FF);

    const dim3 gD(LL / 32, DD / 64);    // GEMM grid, 32x64 tiles, N = 256
    const dim3 gF(LL / 32, FF / 64);    // GEMM grid, N = 2048
    const dim3 gM(MM / 32, DD / 64);    // GEMM grid for memory-side projections
    const dim3 gA(LL / 16, HH);         // attention grid
    const float scale = 0.17677669529663687f;   // 1/sqrt(32)

    // --- self attention ---
    k_gemm<<<gD, blk32, 0, stream>>>(qx, wisb,            b_in_s,        nullptr, Qb, LL, DD, DD, 0);
    k_gemm<<<gD, blk32, 0, stream>>>(qx, wisb + DD*DD,    b_in_s + DD,   nullptr, Kb, LL, DD, DD, 0);
    k_gemm<<<gD, blk32, 0, stream>>>(tg, wisb + 2*DD*DD,  b_in_s + 2*DD, nullptr, Vb, LL, DD, DD, 0);
    k_attn<<<gA, blk32, 0, stream>>>(Qb, Kb, Vb, at, LL, scale);
    k_gemm<<<gD, blk32, 0, stream>>>(at, wosb, b_out_s, t2, nullptr, LL, DD, DD, 0);
    k_add_ln<<<dim3(LL), blk256, 0, stream>>>(tgt, t2, g1, be1, qpos, ln1f, nullptr, qx);

    // --- cross attention ---
    k_gemm<<<gD, blk32, 0, stream>>>(qx, wicb,            b_in_c,        nullptr, Qb, LL, DD, DD, 0);
    k_gemm<<<gM, blk32, 0, stream>>>(mk, wicb + DD*DD,    b_in_c + DD,   nullptr, Kb, MM, DD, DD, 0);
    k_gemm<<<gM, blk32, 0, stream>>>(mm, wicb + 2*DD*DD,  b_in_c + 2*DD, nullptr, Vb, MM, DD, DD, 0);
    k_attn<<<gA, blk32, 0, stream>>>(Qb, Kb, Vb, at, MM, scale);
    k_gemm<<<gD, blk32, 0, stream>>>(at, wocb, b_out_c, t2, nullptr, LL, DD, DD, 0);
    k_add_ln<<<dim3(LL), blk256, 0, stream>>>(ln1f, t2, g2, be2, nullptr, ln2f, tg, nullptr);

    // --- FFN ---
    k_gemm<<<gF, blk32, 0, stream>>>(tg,  w1b, b1, nullptr, ffh, LL, FF, DD, 1);
    k_gemm<<<gD, blk32, 0, stream>>>(ffh, w2b, b2, t2, nullptr, LL, DD, FF, 0);
    k_add_ln<<<dim3(LL), blk256, 0, stream>>>(ln2f, t2, g3, be3, nullptr,
                                              (float*)d_out, nullptr, nullptr);
}
